// AttentivePoolingNetwork_39599598469924
// MI455X (gfx1250) — compile-verified, matrix operand order checked by reading, not run on hardware
//
#include <hip/hip_runtime.h>

typedef __attribute__((ext_vector_type(16))) _Float16 v16h;
typedef __attribute__((ext_vector_type(8)))  _Float16 v8h;
typedef __attribute__((ext_vector_type(8)))  float    v8f;

#define B_   128
#define LQ_  128
#define LA_  512
#define EMB_ 300
#define EP_  320            // EMB padded to multiple of 32
#define F_   400
#define FP_  416            // FILT padded to multiple of 32
#define KW_  3
#define KC_  (KW_ * EP_)    // 960 = conv GEMM K-dim
#define MT_  25             // m-tiles (F_/16)
#define MB_  13             // m-blocks of 2 tiles (last block has 1)

// monotone float <-> orderable-uint mapping (for LDS atomic max on floats)
__device__ __forceinline__ unsigned encf(float f) {
  unsigned u = __float_as_uint(f);
  return (u & 0x80000000u) ? ~u : (u | 0x80000000u);
}
__device__ __forceinline__ float decf(unsigned u) {
  unsigned v = (u & 0x80000000u) ? (u & 0x7fffffffu) : ~u;
  return __uint_as_float(v);
}

// Load one 16-bit WMMA fragment (A-style / Bt-style): row = lane&15,
// K-halves split across half-waves (K+8 / K+24 for lanes 16-31).
__device__ __forceinline__ v16h ldfrag(const _Float16* __restrict__ p) {
  v8h x = *(const v8h*)p;
  v8h y = *(const v8h*)(p + 16);
  v16h r;
#pragma unroll
  for (int i = 0; i < 8; i++) { r[i] = x[i]; r[i + 8] = y[i]; }
  return r;
}

__device__ __forceinline__ v8f wmma16(v16h a, v16h b, v8f c) {
  return __builtin_amdgcn_wmma_f32_16x16x32_f16(false, a, false, b,
                                                (short)0, c, false, false);
}

// ---------------------------------------------------------------------------
// Repack conv weights [F][E][3] -> f16 [F][3*EP] and W [F][F] -> Wt f16 [F][FP]
// (zero-filled pads so WMMA never sees 0 x garbage)
// ---------------------------------------------------------------------------
__global__ void apn_prep(const float* __restrict__ conv_w,
                         const float* __restrict__ W,
                         _Float16* __restrict__ wconv,
                         _Float16* __restrict__ wtT) {
  int idx = blockIdx.x * blockDim.x + threadIdx.x;
  const int N1 = F_ * KC_;
  const int N2 = F_ * FP_;
  if (idx < N1) {
    int f = idx / KC_, k = idx % KC_;
    int kk = k / EP_, e = k % EP_;
    float v = (e < EMB_) ? conv_w[(f * EMB_ + e) * KW_ + kk] : 0.f;
    wconv[idx] = (_Float16)v;
  } else if (idx < N1 + N2) {
    int j = idx - N1;
    int g = j / FP_, f = j % FP_;
    float v = (f < F_) ? W[f * F_ + g] : 0.f;   // Wt[g][f] = W[f][g]
    wtT[j] = (_Float16)v;
  }
}

// ---------------------------------------------------------------------------
// Embedding gather -> f16 [B][L+2][EP] with zero boundary rows (conv halo)
// ---------------------------------------------------------------------------
__global__ void apn_gather(const int* __restrict__ tok,
                           const float* __restrict__ emb,
                           _Float16* __restrict__ out, int L) {
  long idx = (long)blockIdx.x * blockDim.x + threadIdx.x;
  long N = (long)B_ * (L + 2) * EP_;
  if (idx >= N) return;
  int e = (int)(idx % EP_);
  long r = idx / EP_;
  int p = (int)(r % (L + 2));
  int b = (int)(r / (L + 2));
  float v = 0.f;
  if (p > 0 && p <= L && e < EMB_) {
    int t = tok[(long)b * L + (p - 1)];
    v = emb[(long)t * EMB_ + e];
  }
  out[idx] = (_Float16)v;
}

__global__ void apn_zero(uint4* __restrict__ p, long n16) {
  long i = (long)blockIdx.x * blockDim.x + threadIdx.x;
  if (i < n16) p[i] = make_uint4(0u, 0u, 0u, 0u);
}

// ---------------------------------------------------------------------------
// WMMA f16 GEMM, one wave per 32x32 output block (2x2 tiles, register-blocked:
// 8 b128 loads -> 4 WMMAs per k-step). Last m-block may hold a single tile
// (wave-uniform branch; EXEC stays all-ones around every WMMA).
//   A  : f16 [M][K] row-major (lda), per-batch stride strideA (0 = shared)
//   Bt : f16 [N][K] row-major (ldb) == B column-major
//   Out: f16 [N][M] (ldo) -- transposed store, one v8h per lane per tile
// ---------------------------------------------------------------------------
__global__ __launch_bounds__(256) void apn_gemm(
    const _Float16* __restrict__ A, int lda, long strideA,
    const _Float16* __restrict__ Bt, int ldb, long strideB,
    _Float16* __restrict__ Out, int ldo, long strideO,
    const float* __restrict__ bias,
    int mBlocks, int mTiles, int nBlocks, int kSteps) {
  int wave = (int)((blockIdx.x * blockDim.x + threadIdx.x) >> 5);
  int lane = threadIdx.x & 31;
  int lo = lane & 15, hi = lane >> 4;
  int bpb = mBlocks * nBlocks;
  int b = wave / bpb;
  int t = wave % bpb;
  int mB = t / nBlocks, nB = t % nBlocks;
  int m0 = mB * 32, n0 = nB * 32;
  bool hasM1 = (mB * 2 + 1) < mTiles;

  const _Float16* aRow0 = A + b * strideA + (long)(m0 + lo) * lda;
  const _Float16* aRow1 = aRow0 + (long)16 * lda;
  const _Float16* bRow0 = Bt + b * strideB + (long)(n0 + lo) * ldb;
  const _Float16* bRow1 = bRow0 + (long)16 * ldb;

  v8f c00 = {}, c01 = {}, c10 = {}, c11 = {};
  if (hasM1) {
    for (int ks = 0; ks < kSteps; ++ks) {
      int k0 = ks * 32 + hi * 8;
      v16h a0 = ldfrag(aRow0 + k0);
      v16h a1 = ldfrag(aRow1 + k0);
      v16h b0 = ldfrag(bRow0 + k0);
      v16h b1 = ldfrag(bRow1 + k0);
      c00 = wmma16(a0, b0, c00);
      c01 = wmma16(a0, b1, c01);
      c10 = wmma16(a1, b0, c10);
      c11 = wmma16(a1, b1, c11);
    }
  } else {
    for (int ks = 0; ks < kSteps; ++ks) {
      int k0 = ks * 32 + hi * 8;
      v16h a0 = ldfrag(aRow0 + k0);
      v16h b0 = ldfrag(bRow0 + k0);
      v16h b1 = ldfrag(bRow1 + k0);
      c00 = wmma16(a0, b0, c00);
      c01 = wmma16(a0, b1, c01);
    }
  }

  _Float16* oBase = Out + b * strideO;
  // tile (mt, nt): element (m = m0+mt*16+8*hi+r, n = n0+nt*16+lo) -> Out[n][m]
  {
    float bv0[8], bv1[8];
#pragma unroll
    for (int r = 0; r < 8; r++) {
      bv0[r] = bias ? bias[m0 + hi * 8 + r] : 0.f;
      bv1[r] = (bias && hasM1) ? bias[m0 + 16 + hi * 8 + r] : 0.f;
    }
    v8h d00, d01, d10, d11;
#pragma unroll
    for (int r = 0; r < 8; r++) {
      d00[r] = (_Float16)(c00[r] + bv0[r]);
      d01[r] = (_Float16)(c01[r] + bv0[r]);
      d10[r] = (_Float16)(c10[r] + bv1[r]);
      d11[r] = (_Float16)(c11[r] + bv1[r]);
    }
    *(v8h*)(oBase + (long)(n0 + lo) * ldo + m0 + hi * 8) = d00;
    *(v8h*)(oBase + (long)(n0 + 16 + lo) * ldo + m0 + hi * 8) = d01;
    if (hasM1) {
      *(v8h*)(oBase + (long)(n0 + lo) * ldo + m0 + 16 + hi * 8) = d10;
      *(v8h*)(oBase + (long)(n0 + 16 + lo) * ldo + m0 + 16 + hi * 8) = d11;
    }
  }
}

// tanh + row/col max bookkeeping for one 16x16 G tile
__device__ __forceinline__ void tile_max(v8f c, int qBase, int aBase,
                                         int lo, int hi,
                                         unsigned* mQ, unsigned* mA) {
  unsigned colEnc = 0u;
#pragma unroll
  for (int r = 0; r < 8; r++) {
    float v = tanhf(c[r]);
    unsigned u = encf(v);
    colEnc = colEnc > u ? colEnc : u;
    unsigned ru = u;                 // max over 16 n-lanes within half-wave
#pragma unroll
    for (int m = 1; m < 16; m <<= 1) {
      unsigned o = __shfl_xor(ru, m, 32);
      ru = ru > o ? ru : o;
    }
    if (lo == 0) atomicMax(&mQ[qBase + hi * 8 + r], ru);
  }
  atomicMax(&mA[aBase + lo], colEnc);
}

// ---------------------------------------------------------------------------
// Fused attention: per-batch workgroup (8 waves, wave == q-tile).
// G = tanh(Tt x At^T) on WMMA with 2-way N-blocking (A-fragment reused),
// row/col maxes via LDS u32 atomic-max, softmax (tanh in [-1,1] -> no max
// subtraction needed), attentive pooling, cosine.
// ---------------------------------------------------------------------------
__global__ __launch_bounds__(256) void apn_attn(
    const _Float16* __restrict__ Tt,   // [B][LQ][FP]  (W^T Q, transposed)
    const _Float16* __restrict__ At,   // [B][LA][FP]
    const _Float16* __restrict__ Qt,   // [B][LQ][FP]
    float* __restrict__ out) {
  __shared__ unsigned mQ[LQ_];
  __shared__ unsigned mA[LA_];
  __shared__ float roQ[LQ_];
  __shared__ float roA[LA_];
  __shared__ float red[256];
  __shared__ float rQ[F_];
  __shared__ float rA[F_];
  int b = blockIdx.x, tid = threadIdx.x;
  unsigned initE = encf(-2.0f);
  for (int i = tid; i < LA_; i += 256) mA[i] = initE;
  for (int i = tid; i < LQ_; i += 256) mQ[i] = initE;
  __syncthreads();

  int wave = tid >> 5, lane = tid & 31, lo = lane & 15, hi = lane >> 4;
  const _Float16* tB = Tt + (long)b * LQ_ * FP_;
  const _Float16* aB = At + (long)b * LA_ * FP_;
  int qT = wave;                                    // 8 waves == 8 q-tiles
  const _Float16* aRow = tB + (long)(qT * 16 + lo) * FP_;
  for (int aT2 = 0; aT2 < LA_ / 32; ++aT2) {
    const _Float16* bRow0 = aB + (long)(aT2 * 32 + lo) * FP_;
    const _Float16* bRow1 = bRow0 + (long)16 * FP_;
    __builtin_prefetch(bRow0 + (long)32 * FP_, 0, 0);  // next B strip (stays in ws)
    v8f c0 = {}, c1 = {};
    for (int ks = 0; ks < FP_ / 32; ++ks) {
      int k0 = ks * 32 + hi * 8;
      v16h a = ldfrag(aRow + k0);
      c0 = wmma16(a, ldfrag(bRow0 + k0), c0);
      c1 = wmma16(a, ldfrag(bRow1 + k0), c1);
    }
    tile_max(c0, qT * 16, aT2 * 32, lo, hi, mQ, mA);
    tile_max(c1, qT * 16, aT2 * 32 + 16, lo, hi, mQ, mA);
  }
  __syncthreads();

  // softmax over answers
  float s = 0.f;
  for (int i = tid; i < LA_; i += 256) { float e = expf(decf(mA[i])); roA[i] = e; s += e; }
  red[tid] = s;
  for (int st = 128; st > 0; st >>= 1) { __syncthreads(); if (tid < st) red[tid] += red[tid + st]; }
  __syncthreads();
  float sumA = red[0];
  __syncthreads();
  // softmax over questions
  s = 0.f;
  for (int i = tid; i < LQ_; i += 256) { float e = expf(decf(mQ[i])); roQ[i] = e; s += e; }
  red[tid] = s;
  for (int st = 128; st > 0; st >>= 1) { __syncthreads(); if (tid < st) red[tid] += red[tid + st]; }
  __syncthreads();
  float sumQ = red[0];
  __syncthreads();
  for (int i = tid; i < LA_; i += 256) roA[i] /= sumA;
  for (int i = tid; i < LQ_; i += 256) roQ[i] /= sumQ;
  __syncthreads();

  // attentive pooling
  const _Float16* qB = Qt + (long)b * LQ_ * FP_;
  for (int f = tid; f < F_; f += 256) {
    float acc = 0.f;
    for (int j = 0; j < LQ_; j++) acc += (float)qB[(long)j * FP_ + f] * roQ[j];
    rQ[f] = acc;
  }
  for (int f = tid; f < F_; f += 256) {
    float acc = 0.f;
    for (int j = 0; j < LA_; j++) acc += (float)aB[(long)j * FP_ + f] * roA[j];
    rA[f] = acc;
  }
  __syncthreads();

  // cosine similarity
  float dd = 0.f, nq = 0.f, na = 0.f;
  for (int f = tid; f < F_; f += 256) { float x = rQ[f], y = rA[f]; dd += x * y; nq += x * x; na += y * y; }
  red[tid] = dd;
  for (int st = 128; st > 0; st >>= 1) { __syncthreads(); if (tid < st) red[tid] += red[tid + st]; }
  __syncthreads(); float D = red[0]; __syncthreads();
  red[tid] = nq;
  for (int st = 128; st > 0; st >>= 1) { __syncthreads(); if (tid < st) red[tid] += red[tid + st]; }
  __syncthreads(); float NQ = red[0]; __syncthreads();
  red[tid] = na;
  for (int st = 128; st > 0; st >>= 1) { __syncthreads(); if (tid < st) red[tid] += red[tid + st]; }
  __syncthreads(); float NA = red[0];
  if (tid == 0)
    out[b] = D / (fmaxf(sqrtf(NQ), 1e-8f) * fmaxf(sqrtf(NA), 1e-8f));
}

// ---------------------------------------------------------------------------
extern "C" void kernel_launch(void* const* d_in, const int* in_sizes, int n_in,
                              void* d_out, int out_size, void* d_ws, size_t ws_size,
                              hipStream_t stream) {
  (void)in_sizes; (void)n_in; (void)out_size; (void)ws_size;
  const int*   questions = (const int*)d_in[0];
  const int*   answers   = (const int*)d_in[1];
  const float* emb       = (const float*)d_in[2];
  const float* conv_w    = (const float*)d_in[3];
  const float* conv_b    = (const float*)d_in[4];
  const float* W         = (const float*)d_in[5];
  float* out = (float*)d_out;

  char* ws = (char*)d_ws;
  size_t off = 0;
  auto carve = [&](size_t bytes) -> char* {
    char* p = ws + off;
    off = (off + bytes + 255) & ~(size_t)255;
    return p;
  };
  _Float16* wconv = (_Float16*)carve((size_t)F_ * KC_ * 2);
  _Float16* wtT   = (_Float16*)carve((size_t)F_ * FP_ * 2);
  _Float16* embQ  = (_Float16*)carve((size_t)B_ * (LQ_ + 2) * EP_ * 2);
  _Float16* embA  = (_Float16*)carve((size_t)B_ * (LA_ + 2) * EP_ * 2);
  char* zbase = ws + off;                       // Qt/At/Tt zeroed as one span
  _Float16* Qt = (_Float16*)carve((size_t)B_ * LQ_ * FP_ * 2);
  _Float16* At = (_Float16*)carve((size_t)B_ * LA_ * FP_ * 2);
  _Float16* Tt = (_Float16*)carve((size_t)B_ * LQ_ * FP_ * 2);
  size_t zbytes = (size_t)((ws + off) - zbase);

  { int N = F_ * KC_ + F_ * FP_;
    apn_prep<<<(N + 255) / 256, 256, 0, stream>>>(conv_w, W, wconv, wtT); }
  { long N = (long)B_ * (LQ_ + 2) * EP_;
    apn_gather<<<(int)((N + 255) / 256), 256, 0, stream>>>(questions, emb, embQ, LQ_); }
  { long N = (long)B_ * (LA_ + 2) * EP_;
    apn_gather<<<(int)((N + 255) / 256), 256, 0, stream>>>(answers, emb, embA, LA_); }
  { long n16 = (long)(zbytes / 16);
    apn_zero<<<(int)((n16 + 255) / 256), 256, 0, stream>>>((uint4*)zbase, n16); }

  // Conv-as-GEMM: ldb = EP (320) < K (960): im2col rows alias the padded
  // [L+2][EP] embedding layout, so addressing stays perfectly linear.
  apn_gemm<<<(B_ * MB_ * 4) / 8, 256, 0, stream>>>(
      wconv, KC_, 0, embQ, EP_, (long)(LQ_ + 2) * EP_,
      Qt, FP_, (long)LQ_ * FP_, conv_b, MB_, MT_, 4, KC_ / 32);
  apn_gemm<<<(B_ * MB_ * 16) / 8, 256, 0, stream>>>(
      wconv, KC_, 0, embA, EP_, (long)(LA_ + 2) * EP_,
      At, FP_, (long)LA_ * FP_, conv_b, MB_, MT_, 16, KC_ / 32);
  // T = W^T Q  (A = Wt [g][f], Bt = Qt [q][f], Out = Tt [q][g])
  apn_gemm<<<(B_ * MB_ * 4) / 8, 256, 0, stream>>>(
      wtT, FP_, 0, Qt, FP_, (long)LQ_ * FP_,
      Tt, FP_, (long)LQ_ * FP_, nullptr, MB_, MT_, 4, FP_ / 32);

  apn_attn<<<B_, 256, 0, stream>>>(Tt, At, Qt, out);
}
